// ProbSparseSelfAttention_25761213841475
// MI455X (gfx1250) — compile-verified, matrix operand order checked by reading, not run on hardware
//
#include <hip/hip_runtime.h>
#include <hip/hip_bf16.h>

// ProbSparse self-attention (Informer) for MI455X / gfx1250, wave32 + WMMA.
// B=8, L_Q=L_K=4096, D_MODEL=512, H=8, E=64, u = sample_k = int(5*ln(4097)) = 41.
//
// All four dense GEMMs (69 GFLOP total, the dominant cost) run on
// v_wmma_f32_16x16x32_bf16 (bf16 operands, fp32 accumulate). Each wave computes
// a 16x64 output strip: one A fragment feeds 4 WMMAs (weights are L2-resident,
// A panel reused across the 8 waves of a block via L0/L2). Sparse scoring /
// top-k / online-softmax run on VALU (negligible FLOPs, ~4 GFLOP combined).

typedef __attribute__((ext_vector_type(16))) __bf16 v16bf;
typedef __attribute__((ext_vector_type(8)))  float  v8f;

#define NEG_INF (-__builtin_inff())

// ---------------------------------------------------------------------------
// Weight transpose + fp32->bf16:  Wt[n][k] = (bf16) W[k][n]   (512 x 512)
// ---------------------------------------------------------------------------
__global__ __launch_bounds__(256) void wt_kernel(const float* __restrict__ W,
                                                 void* __restrict__ Wt_) {
  __bf16* Wt = (__bf16*)Wt_;
  int idx = blockIdx.x * 256 + threadIdx.x;   // over 512*512
  int n = idx >> 9;
  int k = idx & 511;
  Wt[idx] = (__bf16)W[k * 512 + n];
}

// ---------------------------------------------------------------------------
// WMMA GEMM: C[M x N] = A[M x K] * B[K x N], B given transposed (Bt = N x K,
// row-major bf16). Each wave computes a 16x64 strip (4 n-tiles): one A
// fragment -> 4 WMMAs per 32-wide k-step. A block of 8 waves covers one full
// 16-row mt panel across N=512 (A panel L0/L2-hot).
//   MODE 0: A is fp32 (converted to bf16 in-register); C written bf16 into
//           (B,H,L,E) layout: b=r>>12, l=r&4095, h=c>>6, e=c&63
//   MODE 1: A is bf16; C written fp32 row-major with bias added.
//
// Fragment layout (CDNA5 ISA, 16-bit 16x32 A / 32x16 B):
//   lane: half = lane>>4, m/n = lane&15
//   v16bf element i -> K = 8*half + i      (i in 0..7)
//                      K = 8*half + 8 + i  (i in 8..15)
//   C/D element j -> (m = j + 8*half, n = lane&15)
// ---------------------------------------------------------------------------
template <int MODE>
__global__ __launch_bounds__(256) void wmma_gemm_bf16(
    const void* __restrict__ Aptr, const void* __restrict__ Bt_,
    void* __restrict__ Cptr, const float* __restrict__ bias,
    int Mrows, int N, int K) {
  constexpr int NT = 4;                       // n-tiles per wave (16x64 strip)
  const __bf16* Bt = (const __bf16*)Bt_;
  int lane = threadIdx.x & 31;
  int wave = threadIdx.x >> 5;
  int tiles_n = N >> 4;                       // 32
  int ngroups = tiles_n / NT;                 // 8
  int grp = blockIdx.x * (blockDim.x >> 5) + wave;
  if (grp >= (Mrows >> 4) * ngroups) return;  // wave-uniform: EXEC stays all-1s
  int mt = grp / ngroups;
  int nt0 = (grp - mt * ngroups) * NT;

  int hf = lane >> 4;                // 0/1: which K-half this lane carries
  int mr = mt * 16 + (lane & 15);    // A row this lane feeds
  int nl = lane & 15;

  v8f acc[NT] = {};
  for (int k0 = 0; k0 < K; k0 += 32) {
    int ka = k0 + 8 * hf;
    v16bf a;
    if (MODE == 0) {
      const float* A = (const float*)Aptr + (size_t)mr * K;
#pragma unroll
      for (int i = 0; i < 8; ++i) a[i] = (__bf16)A[ka + i];
#pragma unroll
      for (int i = 0; i < 8; ++i) a[8 + i] = (__bf16)A[ka + 16 + i];
      if (k0 + 32 < K) __builtin_prefetch(A + ka + 32, 0, 1);  // global_prefetch_b8
    } else {
      const __bf16* A = (const __bf16*)Aptr + (size_t)mr * K;
#pragma unroll
      for (int i = 0; i < 8; ++i) a[i] = A[ka + i];
#pragma unroll
      for (int i = 0; i < 8; ++i) a[8 + i] = A[ka + 16 + i];
      if (k0 + 32 < K) __builtin_prefetch(A + ka + 32, 0, 1);
    }
#pragma unroll
    for (int t = 0; t < NT; ++t) {
      const __bf16* Brow = Bt + (size_t)((nt0 + t) * 16 + nl) * K;
      v16bf b;
#pragma unroll
      for (int i = 0; i < 8; ++i) b[i] = Brow[ka + i];
#pragma unroll
      for (int i = 0; i < 8; ++i) b[8 + i] = Brow[ka + 16 + i];
      acc[t] = __builtin_amdgcn_wmma_f32_16x16x32_bf16(
          /*neg_a=*/false, a, /*neg_b=*/false, b,
          /*c_mod=*/(short)0, acc[t], /*reuse_a=*/false, /*reuse_b=*/false);
    }
  }

#pragma unroll
  for (int t = 0; t < NT; ++t) {
#pragma unroll
    for (int j = 0; j < 8; ++j) {
      int r = mt * 16 + j + 8 * hf;
      int c = (nt0 + t) * 16 + nl;
      if (MODE == 0) {
        // scatter into (B, H, L, E) bf16:  L=4096, H=8, E=64
        int bb = r >> 12, l = r & 4095, h = c >> 6, e = c & 63;
        ((__bf16*)Cptr)[((((size_t)bb * 8 + h) << 12) + l) * 64 + e] = (__bf16)acc[t][j];
      } else {
        ((float*)Cptr)[(size_t)r * N + c] = acc[t][j] + bias[c];
      }
    }
  }
}

// ---------------------------------------------------------------------------
// Sample scores: M[b,h,q] = max_s(Q.K_sample) - mean_s(Q.K_sample)
// One wave per (b,h,q); lane owns e = {2*lane, 2*lane+1}; shfl_xor reduce.
// ---------------------------------------------------------------------------
__global__ __launch_bounds__(256) void sample_score_kernel(
    const void* __restrict__ Qb_, const void* __restrict__ Kb_,
    const int* __restrict__ sidx, float* __restrict__ Mbuf, int u_samp) {
  const __bf16* Qb = (const __bf16*)Qb_;
  const __bf16* Kb = (const __bf16*)Kb_;
  int lane = threadIdx.x & 31;
  int wid = blockIdx.x * 8 + (threadIdx.x >> 5);   // = bh*4096 + q
  int q = wid & 4095;
  int bh = wid >> 12;
  const __bf16* qp = Qb + (size_t)wid * 64;
  float q0 = (float)qp[2 * lane], q1 = (float)qp[2 * lane + 1];
  float mx = NEG_INF, sm = 0.f;
  for (int s = 0; s < u_samp; ++s) {
    int kidx = sidx[q * u_samp + s];
    const __bf16* kp = Kb + (((size_t)bh << 12) + kidx) * 64;
    float p = q0 * (float)kp[2 * lane] + q1 * (float)kp[2 * lane + 1];
#pragma unroll
    for (int off = 16; off > 0; off >>= 1) p += __shfl_xor(p, off, 32);
    mx = fmaxf(mx, p);
    sm += p;
  }
  if (lane == 0) Mbuf[wid] = mx - sm / (float)u_samp;
}

// ---------------------------------------------------------------------------
// Iterative top-u argmax per (b,h) over 4096 scores, in LDS.
// ---------------------------------------------------------------------------
__global__ __launch_bounds__(256) void topk_kernel(const float* __restrict__ Mbuf,
                                                   int* __restrict__ top_idx, int u) {
  __shared__ float vals[4096];
  __shared__ float rv[256];
  __shared__ int   ri[256];
  int bh = blockIdx.x;
  const float* src = Mbuf + (size_t)bh * 4096;
  for (int i = threadIdx.x; i < 4096; i += 256) vals[i] = src[i];
  __syncthreads();
  for (int t = 0; t < u; ++t) {
    float best = NEG_INF; int bi = 0;
    for (int i = threadIdx.x; i < 4096; i += 256) {
      float v = vals[i];
      if (v > best) { best = v; bi = i; }
    }
    rv[threadIdx.x] = best; ri[threadIdx.x] = bi;
    __syncthreads();
    for (int s = 128; s > 0; s >>= 1) {
      if (threadIdx.x < s && rv[threadIdx.x + s] > rv[threadIdx.x]) {
        rv[threadIdx.x] = rv[threadIdx.x + s];
        ri[threadIdx.x] = ri[threadIdx.x + s];
      }
      __syncthreads();
    }
    if (threadIdx.x == 0) {
      top_idx[bh * u + t] = ri[0];
      vals[ri[0]] = NEG_INF;
    }
    __syncthreads();
  }
}

// ---------------------------------------------------------------------------
// vmean[b,h,e] = mean_l V[b,h,l,e]
// ---------------------------------------------------------------------------
__global__ __launch_bounds__(256) void vmean_kernel(const void* __restrict__ Vb_,
                                                    float* __restrict__ vmean) {
  const __bf16* Vb = (const __bf16*)Vb_;
  __shared__ float part[256];
  int bh = blockIdx.x;
  int e = threadIdx.x & 63;
  int chunk = threadIdx.x >> 6;   // 4 chunks of 1024 rows
  const __bf16* base = Vb + (((size_t)bh) << 12) * 64;
  float s = 0.f;
  for (int l = chunk * 1024; l < (chunk + 1) * 1024; ++l)
    s += (float)base[(size_t)l * 64 + e];
  part[threadIdx.x] = s;
  __syncthreads();
  if (chunk == 0)
    vmean[bh * 64 + e] =
        (part[e] + part[64 + e] + part[128 + e] + part[192 + e]) * (1.0f / 4096.0f);
}

// ---------------------------------------------------------------------------
// Broadcast-fill ctx[b, l, h*64+e] = vmean[b,h,e]   (bf16, 16.7M elems)
// ---------------------------------------------------------------------------
__global__ __launch_bounds__(256) void fill_ctx_kernel(const float* __restrict__ vmean,
                                                       void* __restrict__ ctx_) {
  __bf16* ctx = (__bf16*)ctx_;
  size_t idx = (size_t)blockIdx.x * 256 + threadIdx.x;   // < 8*4096*512
  int col = (int)(idx & 511);
  int b = (int)(idx >> 21);                   // idx / (4096*512)
  int h = col >> 6, e = col & 63;
  ctx[idx] = (__bf16)vmean[((b << 3) + h) * 64 + e];
}

// ---------------------------------------------------------------------------
// Sparse attention rows: for each of the u active queries per (b,h), online
// softmax over all 4096 keys, ctx = attn @ V, scattered over the broadcast fill.
// One wave per (b,h,iu); lane owns e = {2*lane, 2*lane+1}.
// ---------------------------------------------------------------------------
__global__ __launch_bounds__(256) void prob_attn_kernel(
    const void* __restrict__ Qb_, const void* __restrict__ Kb_,
    const void* __restrict__ Vb_, const int* __restrict__ top_idx,
    void* __restrict__ ctx_, int u) {
  const __bf16* Qb = (const __bf16*)Qb_;
  const __bf16* Kb = (const __bf16*)Kb_;
  const __bf16* Vb = (const __bf16*)Vb_;
  __bf16* ctx = (__bf16*)ctx_;
  int lane = threadIdx.x & 31;
  int wid = blockIdx.x * 8 + (threadIdx.x >> 5);
  if (wid >= 64 * u) return;
  int bh = wid / u;
  int iu = wid - bh * u;
  int qi = top_idx[bh * u + iu];

  const __bf16* qp = Qb + (((size_t)bh << 12) + qi) * 64;
  const float scale = 0.125f;   // 1/sqrt(64)
  float q0 = (float)qp[2 * lane] * scale, q1 = (float)qp[2 * lane + 1] * scale;
  const __bf16* kb = Kb + (((size_t)bh << 12)) * 64;
  const __bf16* vb = Vb + (((size_t)bh << 12)) * 64;

  float m = NEG_INF, l = 0.f, a0 = 0.f, a1 = 0.f;
  for (int k = 0; k < 4096; ++k) {
    float s = q0 * (float)kb[(size_t)k * 64 + 2 * lane] +
              q1 * (float)kb[(size_t)k * 64 + 2 * lane + 1];
#pragma unroll
    for (int off = 16; off > 0; off >>= 1) s += __shfl_xor(s, off, 32);
    float nm = fmaxf(m, s);
    float corr = __expf(m - nm);
    float p = __expf(s - nm);
    l = l * corr + p;
    a0 = a0 * corr + p * (float)vb[(size_t)k * 64 + 2 * lane];
    a1 = a1 * corr + p * (float)vb[(size_t)k * 64 + 2 * lane + 1];
    m = nm;
  }
  int b = bh >> 3, h = bh & 7;
  __bf16* out = ctx + (((size_t)b << 12) + qi) * 512 + h * 64;
  float inv = 1.0f / l;
  out[2 * lane] = (__bf16)(a0 * inv);
  out[2 * lane + 1] = (__bf16)(a1 * inv);
}

// ---------------------------------------------------------------------------
extern "C" void kernel_launch(void* const* d_in, const int* in_sizes, int n_in,
                              void* d_out, int out_size, void* d_ws, size_t ws_size,
                              hipStream_t stream) {
  const float* query    = (const float*)d_in[0];
  const float* key_in   = (const float*)d_in[1];
  const float* value    = (const float*)d_in[2];
  const float* W_Q      = (const float*)d_in[3];
  const float* W_K      = (const float*)d_in[4];
  const float* W_V      = (const float*)d_in[5];
  const float* W_out    = (const float*)d_in[6];
  const float* b_out    = (const float*)d_in[7];
  const int*   sample_i = (const int*)d_in[8];
  (void)in_sizes; (void)n_in; (void)out_size; (void)ws_size;

  constexpr int Bv = 8, L = 4096, D = 512, U = 41;  // U = int(5*ln(4097)) = 41
  constexpr int Mrows = Bv * L;                     // 32768

  char* ws = (char*)d_ws;
  size_t off = 0;
  auto take = [&](size_t bytes) -> void* {
    void* p = ws + off;
    off += (bytes + 255) & ~(size_t)255;
    return p;
  };
  void*  WqT   = take((size_t)D * D * 2);
  void*  WkT   = take((size_t)D * D * 2);
  void*  WvT   = take((size_t)D * D * 2);
  void*  WoT   = take((size_t)D * D * 2);
  void*  Qb    = take((size_t)Mrows * D * 2);
  void*  Kb    = take((size_t)Mrows * D * 2);
  void*  Vb    = take((size_t)Mrows * D * 2);
  float* Mbuf  = (float*)take((size_t)Bv * 8 * L * 4);
  int*   topIx = (int*)take((size_t)Bv * 8 * U * 4);
  float* vmean = (float*)take((size_t)Bv * 8 * 64 * 4);
  void*  ctx   = take((size_t)Mrows * D * 2);

  // 1) weights -> bf16, transposed
  wt_kernel<<<(D * D) / 256, 256, 0, stream>>>(W_Q, WqT);
  wt_kernel<<<(D * D) / 256, 256, 0, stream>>>(W_K, WkT);
  wt_kernel<<<(D * D) / 256, 256, 0, stream>>>(W_V, WvT);
  wt_kernel<<<(D * D) / 256, 256, 0, stream>>>(W_out, WoT);

  // 2) projections (WMMA): 16x64 strip per wave -> 16384 waves -> 2048 blocks
  int gemm_blocks = (Mrows / 16) * (D / 64) / 8;
  wmma_gemm_bf16<0><<<gemm_blocks, 256, 0, stream>>>(query,  WqT, Qb, nullptr, Mrows, D, D);
  wmma_gemm_bf16<0><<<gemm_blocks, 256, 0, stream>>>(key_in, WkT, Kb, nullptr, Mrows, D, D);
  wmma_gemm_bf16<0><<<gemm_blocks, 256, 0, stream>>>(value,  WvT, Vb, nullptr, Mrows, D, D);

  // 3) sampled sparsity scores: 262144 waves -> 32768 blocks
  sample_score_kernel<<<(Bv * 8 * L) / 8, 256, 0, stream>>>(Qb, Kb, sample_i, Mbuf, U);

  // 4) top-41 per (b,h)
  topk_kernel<<<Bv * 8, 256, 0, stream>>>(Mbuf, topIx, U);

  // 5) V mean, 6) broadcast fill, 7) sparse attention scatter
  vmean_kernel<<<Bv * 8, 256, 0, stream>>>(Vb, vmean);
  fill_ctx_kernel<<<((size_t)Mrows * D) / 256, 256, 0, stream>>>(vmean, ctx);
  prob_attn_kernel<<<(Bv * 8 * U + 7) / 8, 256, 0, stream>>>(Qb, Kb, Vb, topIx, ctx, U);

  // 8) output projection (WMMA) + bias -> fp32 d_out
  wmma_gemm_bf16<1><<<gemm_blocks, 256, 0, stream>>>(ctx, WoT, d_out, b_out, Mrows, D, D);
}
